// MultiHeadAttention_45251775430983
// MI455X (gfx1250) — compile-verified, matrix operand order checked by reading, not run on hardware
//
#include <hip/hip_runtime.h>

// ---------------------------------------------------------------------------
// MultiHeadAttention (B=1, T=4096, D=768, H=12, HD=64) for gfx1250 (MI455X).
// bf16 WMMA + fp32 accumulate everywhere; block-cooperative flash attention:
//   - block = 4 waves = 64 query rows, all waves share one K/V key stream
//   - K/V tiles double-buffered in LDS, staged with async global->LDS loads
//     (ASYNCcnt) when available; 4x less L2 traffic than per-wave streaming
//   - softmax in exp2 domain (Q pre-scaled by 0.125*log2e at projection)
//   - row-sum via WMMA against an all-ones B fragment
//   - only the diagonal chunk runs the masked path
// ---------------------------------------------------------------------------

typedef __bf16 bf16;
typedef __attribute__((ext_vector_type(16))) __bf16 v16bf;
typedef __attribute__((ext_vector_type(8)))  __bf16 v8bf;
typedef __attribute__((ext_vector_type(8)))  float  v8f;
typedef __attribute__((ext_vector_type(4)))  int    v4i;

// address-space-qualified pointer element types for the async-LDS builtin
typedef __attribute__((address_space(1))) v4i g_v4i;   // global
typedef __attribute__((address_space(3))) v4i l_v4i;   // LDS

static constexpr int Tn = 4096;
static constexpr int Dn = 768;
static constexpr int Hn = 12;
static constexpr int HDn = 64;

union frag_u { v16bf v; v8bf h[2]; };

// 16x32 bf16 WMMA fragment (A rows=M / B rows=N from transposed storage).
// ISA 7.12.2: r = lane&15 row; kh = lane>>4; element e -> K = 16*(e>>3)+8*kh+(e&7)
static __device__ __forceinline__ v16bf load_frag(const bf16* __restrict__ base,
                                                  int stride, int lane) {
  const int r  = lane & 15;
  const int kh = (lane >> 4) & 1;
  const bf16* p = base + r * stride + kh * 8;
  frag_u f;
  f.h[0] = *(const v8bf*)(p);
  f.h[1] = *(const v8bf*)(p + 16);
  return f.v;
}

static __device__ __forceinline__ v8f wmma_bf16(v16bf a, v16bf b, v8f c) {
  return __builtin_amdgcn_wmma_f32_16x16x32_bf16(false, a, false, b,
                                                 (short)0, c, false, false);
}

static __device__ __forceinline__ float fast_exp2(float x) {
  return __builtin_amdgcn_exp2f(x);
}

// ---- async global -> LDS staging (CDNA5 async path, ASYNCcnt) -------------
static __device__ __forceinline__ void async_copy_b128(bf16* lds,
                                                       const bf16* g) {
#if __has_builtin(__builtin_amdgcn_global_load_async_to_lds_b128)
  __builtin_amdgcn_global_load_async_to_lds_b128(
      (g_v4i*)g, (l_v4i*)lds, 0, 0);
#else
  *(v4i*)lds = *(const v4i*)g;  // global_load_b128 + ds_store_b128 fallback
#endif
}

static __device__ __forceinline__ void wait_async0() {
#if __has_builtin(__builtin_amdgcn_s_wait_asynccnt)
  __builtin_amdgcn_s_wait_asynccnt(0);
#else
  __asm__ volatile("s_wait_asynccnt 0" ::: "memory");
#endif
}

// ---------------------------------------------------------------------------
// Conversion kernels
// ---------------------------------------------------------------------------
__global__ void cvt_bf16_kernel(const float* __restrict__ in,
                                bf16* __restrict__ out, int n) {
  int i = blockIdx.x * blockDim.x + threadIdx.x;
  if (i < n) out[i] = (bf16)in[i];
}

__global__ void cvt_transpose_kernel(const float* __restrict__ W,
                                     bf16* __restrict__ Wt, int dim) {
  int i = blockIdx.x * blockDim.x + threadIdx.x;
  if (i < dim * dim) {
    int n = i / dim;
    int k = i - n * dim;
    Wt[i] = (bf16)W[k * dim + n];
  }
}

// ---------------------------------------------------------------------------
// Generic bf16 GEMM: C[M,N] = A[M,K] * Bt[N,K]^T   (16x64 tile per wave)
// MODE 0: bf16 row-major store, scaled by out_scale (ldc = N)
// MODE 1: bf16 transposed store C[n*ldc + m]       (ldc = M)  -- V^T
// MODE 2: f32 row-major + bias                     (ldc = N)  -- out proj
// ---------------------------------------------------------------------------
template <int MODE>
__global__ __launch_bounds__(128)
void gemm_bf16_kernel(const bf16* __restrict__ A, int lda,
                      const bf16* __restrict__ Bt, int ldb,
                      void* __restrict__ Cout, int ldc,
                      const float* __restrict__ bias, float out_scale,
                      int M, int N, int K) {
  const int lane = threadIdx.x & 31;
  const int wave = threadIdx.x >> 5;
  const int job  = blockIdx.x * 4 + wave;
  const int mtiles = M >> 4;
  const int mt = job % mtiles;
  const int nt = job / mtiles;
  const int m0 = mt << 4;
  const int n0 = nt << 6;
  if (n0 >= N) return;

  v8f acc[4] = {};
  for (int k0 = 0; k0 < K; k0 += 32) {
    v16bf a = load_frag(A + m0 * lda + k0, lda, lane);
#pragma unroll
    for (int t = 0; t < 4; ++t) {
      v16bf b = load_frag(Bt + (n0 + t * 16) * ldb + k0, ldb, lane);
      acc[t] = wmma_bf16(a, b, acc[t]);
    }
  }

  const int nl  = lane & 15;
  const int mhi = (lane >> 4) << 3;
  if (MODE == 0) {
    bf16* C = (bf16*)Cout;
#pragma unroll
    for (int t = 0; t < 4; ++t)
#pragma unroll
      for (int r = 0; r < 8; ++r)
        C[(m0 + mhi + r) * ldc + n0 + t * 16 + nl] = (bf16)(acc[t][r] * out_scale);
  } else if (MODE == 1) {
    bf16* C = (bf16*)Cout;
#pragma unroll
    for (int t = 0; t < 4; ++t)
#pragma unroll
      for (int r = 0; r < 8; ++r)
        C[(n0 + t * 16 + nl) * ldc + m0 + mhi + r] = (bf16)acc[t][r];
  } else {
    float* C = (float*)Cout;
#pragma unroll
    for (int t = 0; t < 4; ++t) {
      float b = bias ? bias[n0 + t * 16 + nl] : 0.0f;
#pragma unroll
      for (int r = 0; r < 8; ++r)
        C[(m0 + mhi + r) * ldc + n0 + t * 16 + nl] = acc[t][r] + b;
    }
  }
}

// ---------------------------------------------------------------------------
// Stage one 64-key chunk of K (rows=key, 64 hd) and V^T (rows=hd, 64 keys)
// into LDS; 128 threads x (4+4) b128 segments.
// ---------------------------------------------------------------------------
static __device__ __forceinline__ void stage_kv(
    const bf16* __restrict__ Kh, const bf16* __restrict__ Vh,
    bf16 (*__restrict__ Ksb)[64], bf16 (*__restrict__ Vsb)[64],
    int kb, int tid) {
#pragma unroll
  for (int i = 0; i < 4; ++i) {
    const int s   = tid + i * 128;   // 0..511 segments of 8 bf16
    const int row = s >> 3;
    const int off = (s & 7) * 8;
    async_copy_b128(&Ksb[row][off], Kh + (kb + row) * Dn + off);
    async_copy_b128(&Vsb[row][off], Vh + row * Tn + kb + off);
  }
}

// ---------------------------------------------------------------------------
// One 64-key flash-attention chunk, K/V fragments from LDS.
// ---------------------------------------------------------------------------
template <bool MASKED>
static __device__ __forceinline__ void attn_chunk64(
    const bf16 (*__restrict__ Ksb)[64], const bf16 (*__restrict__ Vsb)[64],
    bf16 (*__restrict__ plds)[64],
    const v16bf& qf0, const v16bf& qf1, const v16bf& ones,
    v8f (&acc)[4], float (&mrow)[8], float (&lrow)[8],
    int kb, int q0w, int lane) {
  const int nl  = lane & 15;
  const int mhi = (lane >> 4) << 3;

  // ---- S tiles (2 WMMAs each) ----
  v8f st[4];
#pragma unroll
  for (int t = 0; t < 4; ++t) {
    v16bf kf0 = load_frag(&Ksb[t * 16][0],  64, lane);
    v16bf kf1 = load_frag(&Ksb[t * 16][32], 64, lane);
    v8f s = {};
    s = wmma_bf16(qf0, kf0, s);
    s = wmma_bf16(qf1, kf1, s);
    if (MASKED) {
      const int key = kb + t * 16 + nl;
#pragma unroll
      for (int r = 0; r < 8; ++r)
        s[r] = (key <= q0w + mhi + r) ? s[r] : -1e30f;
    }
    st[t] = s;
  }

  // ---- row max: in-lane over 4 tiles, then 16-lane butterfly ----
  float corr[8];
#pragma unroll
  for (int r = 0; r < 8; ++r) {
    float mx = fmaxf(fmaxf(st[0][r], st[1][r]), fmaxf(st[2][r], st[3][r]));
#pragma unroll
    for (int off = 1; off < 16; off <<= 1)
      mx = fmaxf(mx, __shfl_xor(mx, off, 32));
    const float mn = fmaxf(mrow[r], mx);
    corr[r] = fast_exp2(mrow[r] - mn);
    mrow[r] = mn;
  }

  // ---- P = exp2(S - m): C-layout -> A-layout via per-wave LDS tile ----
#pragma unroll
  for (int t = 0; t < 4; ++t)
#pragma unroll
    for (int r = 0; r < 8; ++r)
      plds[mhi + r][t * 16 + nl] = (bf16)fast_exp2(st[t][r] - mrow[r]);
  __asm__ volatile("s_wait_dscnt 0" ::: "memory");
  v16bf pf0 = load_frag(&plds[0][0],  64, lane);
  v16bf pf1 = load_frag(&plds[0][32], 64, lane);

  // ---- row sum via ones-WMMA ----
  v8f ls = {};
  ls = wmma_bf16(pf0, ones, ls);
  ls = wmma_bf16(pf1, ones, ls);
#pragma unroll
  for (int r = 0; r < 8; ++r) lrow[r] = lrow[r] * corr[r] + ls[r];

  // ---- rescale accumulator, ctx += P * V ----
#pragma unroll
  for (int t2 = 0; t2 < 4; ++t2) {
#pragma unroll
    for (int r = 0; r < 8; ++r) acc[t2][r] *= corr[r];
    v16bf vf0 = load_frag(&Vsb[t2 * 16][0],  64, lane);
    v16bf vf1 = load_frag(&Vsb[t2 * 16][32], 64, lane);
    acc[t2] = wmma_bf16(pf0, vf0, acc[t2]);
    acc[t2] = wmma_bf16(pf1, vf1, acc[t2]);
  }
  __asm__ volatile("" ::: "memory");
}

__global__ __launch_bounds__(128)
void attn_kernel(const bf16* __restrict__ Q,
                 const bf16* __restrict__ Km,
                 const bf16* __restrict__ Vt,
                 bf16* __restrict__ ctx) {
  const int tid  = threadIdx.x;
  const int lane = tid & 31;
  const int wave = tid >> 5;
  const int h     = blockIdx.y;
  const int q0blk = blockIdx.x << 6;       // 64 query rows per block
  const int q0w   = q0blk + (wave << 4);   // this wave's 16 rows
  const int hoff  = h * HDn;

  __shared__ bf16 Ks[2][64][64];        // 16 KB double-buffered K tile
  __shared__ bf16 Vs[2][64][64];        // 16 KB double-buffered V^T tile
  __shared__ bf16 plds_all[4][16][64];  //  8 KB per-wave P transpose tiles
  bf16(*plds)[64] = plds_all[wave];

  const bf16* Kh = Km + hoff;       // [T][D] + head column offset
  const bf16* Vh = Vt + hoff * Tn;  // [D][T] + head row offset

  const v16bf qf0 = load_frag(Q + q0w * Dn + hoff,      Dn, lane);
  const v16bf qf1 = load_frag(Q + q0w * Dn + hoff + 32, Dn, lane);

  v16bf ones;
#pragma unroll
  for (int e = 0; e < 16; ++e) ones[e] = (bf16)1.0f;

  v8f acc[4] = {};
  float mrow[8], lrow[8];
#pragma unroll
  for (int r = 0; r < 8; ++r) { mrow[r] = -1e30f; lrow[r] = 0.0f; }

  const int nch = (q0blk >> 6) + 1;  // 64-key chunks incl. masked diagonal

  stage_kv(Kh, Vh, Ks[0], Vs[0], 0, tid);
  wait_async0();
  __syncthreads();

  for (int c = 0; c < nch; ++c) {
    const int kb = c << 6;
    if (c + 1 < nch)  // prefetch next chunk into the other buffer
      stage_kv(Kh, Vh, Ks[(c + 1) & 1], Vs[(c + 1) & 1], kb + 64, tid);
    if (c + 1 == nch)
      attn_chunk64<true>(Ks[c & 1], Vs[c & 1], plds, qf0, qf1, ones,
                         acc, mrow, lrow, kb, q0w, lane);
    else
      attn_chunk64<false>(Ks[c & 1], Vs[c & 1], plds, qf0, qf1, ones,
                          acc, mrow, lrow, kb, q0w, lane);
    wait_async0();
    __syncthreads();
  }

  const int nl  = lane & 15;
  const int mhi = (lane >> 4) << 3;
#pragma unroll
  for (int r = 0; r < 8; ++r) {
    const float inv = 1.0f / lrow[r];
#pragma unroll
    for (int t = 0; t < 4; ++t)
      ctx[(q0w + mhi + r) * Dn + hoff + t * 16 + nl] = (bf16)(acc[t][r] * inv);
  }
}

// ---------------------------------------------------------------------------
// Launch
// ---------------------------------------------------------------------------
static inline size_t align256(size_t x) { return (x + 255) & ~size_t(255); }

extern "C" void kernel_launch(void* const* d_in, const int* in_sizes, int n_in,
                              void* d_out, int out_size, void* d_ws,
                              size_t ws_size, hipStream_t stream) {
  const float* x  = (const float*)d_in[0];
  const float* Wq = (const float*)d_in[1];
  const float* Wk = (const float*)d_in[2];
  const float* Wv = (const float*)d_in[3];
  const float* Wo = (const float*)d_in[4];
  const float* bo = (const float*)d_in[5];
  float* out = (float*)d_out;

  char* ws = (char*)d_ws;
  const size_t szTD = (size_t)Tn * Dn * sizeof(bf16);
  const size_t szDD = (size_t)Dn * Dn * sizeof(bf16);
  bf16* Xb  = (bf16*)ws;  ws += align256(szTD);
  bf16* Wqt = (bf16*)ws;  ws += align256(szDD);
  bf16* Wkt = (bf16*)ws;  ws += align256(szDD);
  bf16* Wvt = (bf16*)ws;  ws += align256(szDD);
  bf16* Wot = (bf16*)ws;  ws += align256(szDD);
  bf16* Qb  = (bf16*)ws;  ws += align256(szTD);
  bf16* Kb  = (bf16*)ws;  ws += align256(szTD);
  bf16* Vtb = (bf16*)ws;  ws += align256(szTD);  // [D][T]
  bf16* Cxb = (bf16*)ws;  ws += align256(szTD);

  {
    int n = Tn * Dn;
    cvt_bf16_kernel<<<(n + 255) / 256, 256, 0, stream>>>(x, Xb, n);
    int m = Dn * Dn;
    int g = (m + 255) / 256;
    cvt_transpose_kernel<<<g, 256, 0, stream>>>(Wq, Wqt, Dn);
    cvt_transpose_kernel<<<g, 256, 0, stream>>>(Wk, Wkt, Dn);
    cvt_transpose_kernel<<<g, 256, 0, stream>>>(Wv, Wvt, Dn);
    cvt_transpose_kernel<<<g, 256, 0, stream>>>(Wo, Wot, Dn);
  }

  const int gemm_blocks = (Tn / 16) * (Dn / 64) / 4;
  const float qscale = 0.125f * 1.44269504088896340736f;  // 1/sqrt(HD)*log2e
  gemm_bf16_kernel<0><<<gemm_blocks, 128, 0, stream>>>(
      Xb, Dn, Wqt, Dn, Qb, Dn, nullptr, qscale, Tn, Dn, Dn);
  gemm_bf16_kernel<0><<<gemm_blocks, 128, 0, stream>>>(
      Xb, Dn, Wkt, Dn, Kb, Dn, nullptr, 1.0f, Tn, Dn, Dn);
  gemm_bf16_kernel<1><<<gemm_blocks, 128, 0, stream>>>(
      Xb, Dn, Wvt, Dn, Vtb, Tn, nullptr, 1.0f, Tn, Dn, Dn);

  attn_kernel<<<dim3(Tn / 64, Hn), 128, 0, stream>>>(Qb, Kb, Vtb, Cxb);

  gemm_bf16_kernel<2><<<gemm_blocks, 128, 0, stream>>>(
      Cxb, Dn, Wot, Dn, out, Dn, bo, 1.0f, Tn, Dn, Dn);
}
